// Cell_59382217834843
// MI455X (gfx1250) — compile-verified, matrix operand order checked by reading, not run on hardware
//
#include <hip/hip_runtime.h>
#include <hip/hip_bf16.h>

// ---------------------------------------------------------------------------
// NAS-RNN cell for gfx1250 (MI455X), fully register-resident recurrence.
//   H=1024, 2H=2048, N(batch)=256, STEPS=8, L=64
// - Math: v_wmma_f32_16x16x32_bf16 (bf16 in, f32 accumulate), weights bf16 in L2.
// - Each wave computes a 16M x 32N c-tile pair AND its matching hh-tile pair
//   (cols j and j+1024): gating + the s/acc recurrence stay in registers.
// - B operand staged global->LDS with global_load_async_to_lds_b128
//   (ASYNCcnt, double buffered, 64-wide K chunks, 144B padded rows ->
//   conflict-free ds_load_b128), shared by all 8 waves of the block.
// - 8 grid barriers per timestep (atomic ticket barrier, 64 co-resident blocks).
// ---------------------------------------------------------------------------

#define H      1024
#define H2     2048
#define NB     256
#define STEPS  8
#define SEQL   64
#define NH     (NB * H)

#define GRID_BLOCKS   64     // 2 m-slabs (128 rows) x 32 j-strips (32 cols)
#define BLOCK_THREADS 256    // 8 waves, all sharing one j-strip

// LDS staging: 2 buffers x 64 rows x 128B data (K chunk = 64 bf16), rows padded
// to 144B so the 16 row-reads of a fragment start in 16 distinct bank groups.
#define ROW_BYTES 144
#define BUF_BYTES (64 * ROW_BYTES)     // 9216 B per buffer

typedef __attribute__((ext_vector_type(16))) __bf16 v16bf;
typedef __attribute__((ext_vector_type(8)))  float  v8f;

union FragB {
  v16bf v;
  uint4 u[2];
};

__device__ __forceinline__ __bf16 to_bf16(float f) {
  unsigned u = __float_as_uint(f);
  u = (u + 0x7fffu + ((u >> 16) & 1u)) >> 16;     // RNE
  unsigned short h = (unsigned short)u;
  __bf16 b;
  __builtin_memcpy(&b, &h, sizeof(b));
  return b;
}

__device__ __forceinline__ float sigmoid_f(float x) {
  return 1.0f / (1.0f + __expf(-x));
}

// Grid-wide barrier (monotonic ticket, agent scope). 64 blocks are co-resident.
__device__ __forceinline__ void grid_sync(unsigned int* bar) {
  __threadfence();
  __syncthreads();
  if (threadIdx.x == 0) {
    unsigned int t = __hip_atomic_fetch_add(bar, 1u, __ATOMIC_ACQ_REL,
                                            __HIP_MEMORY_SCOPE_AGENT);
    unsigned int target = t - (t % GRID_BLOCKS) + GRID_BLOCKS;
    while (__hip_atomic_load(bar, __ATOMIC_ACQUIRE,
                             __HIP_MEMORY_SCOPE_AGENT) < target) {
      __builtin_amdgcn_s_sleep(1);
    }
  }
  __syncthreads();
  __threadfence();
}

// CDNA5 async global->LDS copy (16B per lane), tracked by ASYNCcnt.
__device__ __forceinline__ void async_copy_b128(unsigned lds_addr,
                                                const __bf16* gaddr) {
  asm volatile("global_load_async_to_lds_b128 %0, %1, off"
               :: "v"(lds_addr), "v"(gaddr) : "memory");
}
__device__ __forceinline__ void wait_async_le2() {
  asm volatile("s_wait_asynccnt 0x2" ::: "memory");
}
__device__ __forceinline__ void wait_async_le0() {
  asm volatile("s_wait_asynccnt 0x0" ::: "memory");
}

// ---------------------------------------------------------------------------
// One GEMM stage. Block output: 128M x {cols [j0,j0+32) and [H+j0,H+j0+32)}.
// A[256 x K] bf16 (lda), BT[2048 n][K k] bf16 (transposed weights).
// LDS rows 0..31 = BT rows j0+r (c gates), rows 32..63 = BT rows H+j0+r (hh).
// Producer: thread t owns row t>>2, 16B chunk t&3; issues 2 async copies per
// 64-wide K chunk. Consumers ds_load fragments (2 k-steps, 8 WMMAs / barrier).
// ---------------------------------------------------------------------------
__device__ __forceinline__ void gemm_stage(
    const __bf16* __restrict__ A, int lda, int K,
    const __bf16* __restrict__ BT,
    unsigned char* smem, int j0, int m0,
    v8f& c0, v8f& c1, v8f& hh0, v8f& hh1) {
  const int lane  = threadIdx.x & 31;
  const int l15   = lane & 15;
  const int khalf = lane >> 4;                 // 0/1: which 8-elem K chunk

  // producer indices (two 16B async copies per thread per 64-K chunk)
  const int prow = threadIdx.x >> 2;           // 0..63
  const int pchk = threadIdx.x & 3;            // 0..3
  const int pn   = (prow < 32) ? (j0 + prow) : (H + j0 + (prow - 32));
  const __bf16* gsrc = BT + (size_t)pn * K + pchk * 8;
  const unsigned lds_dst =
      (unsigned)(size_t)smem + (unsigned)(prow * ROW_BYTES + pchk * 16);

  // consumer byte offsets within one LDS buffer (k-step adds +64)
  const unsigned rb_c0 = (unsigned)((0  + l15) * ROW_BYTES + khalf * 16);
  const unsigned rb_c1 = (unsigned)((16 + l15) * ROW_BYTES + khalf * 16);
  const unsigned rb_h0 = (unsigned)((32 + l15) * ROW_BYTES + khalf * 16);
  const unsigned rb_h1 = (unsigned)((48 + l15) * ROW_BYTES + khalf * 16);

  const __bf16* ap = A + (size_t)(m0 + l15) * lda + khalf * 8;

  c0 = v8f{}; c1 = v8f{}; hh0 = v8f{}; hh1 = v8f{};

  // prime buffer 0 with K chunk [0,64)
  async_copy_b128(lds_dst,      gsrc);
  async_copy_b128(lds_dst + 64, gsrc + 32);

  for (int kb = 0; kb < K; kb += 64) {
    const unsigned cur = ((unsigned)kb >> 6) & 1u;
    if (kb + 64 < K) {
      const unsigned nxt = (cur ^ 1u) * (unsigned)BUF_BYTES;
      async_copy_b128(lds_dst + nxt,      gsrc + kb + 64);
      async_copy_b128(lds_dst + nxt + 64, gsrc + kb + 96);
      wait_async_le2();                        // my 2 copies into `cur` landed
    } else {
      wait_async_le0();
    }
    __syncthreads();                           // buffer `cur` ready block-wide

    __builtin_prefetch(ap + kb + 256, 0, 1);   // A stream -> global_prefetch_b8
    const unsigned char* buf = smem + cur * (unsigned)BUF_BYTES;

#pragma unroll
    for (int ks = 0; ks < 2; ++ks) {           // two 32-wide k-steps
      const int koff = kb + ks * 32;
      FragB a;
      a.u[0] = *(const uint4*)(ap + koff);
      a.u[1] = *(const uint4*)(ap + koff + 16);

      const unsigned sb = (unsigned)(ks * 64);
      FragB bc0, bc1, bh0, bh1;
      bc0.u[0] = *(const uint4*)(buf + rb_c0 + sb);
      bc0.u[1] = *(const uint4*)(buf + rb_c0 + sb + 32);
      bc1.u[0] = *(const uint4*)(buf + rb_c1 + sb);
      bc1.u[1] = *(const uint4*)(buf + rb_c1 + sb + 32);
      bh0.u[0] = *(const uint4*)(buf + rb_h0 + sb);
      bh0.u[1] = *(const uint4*)(buf + rb_h0 + sb + 32);
      bh1.u[0] = *(const uint4*)(buf + rb_h1 + sb);
      bh1.u[1] = *(const uint4*)(buf + rb_h1 + sb + 32);

      c0  = __builtin_amdgcn_wmma_f32_16x16x32_bf16(false, a.v, false, bc0.v,
                                                    (short)0, c0,  false, false);
      c1  = __builtin_amdgcn_wmma_f32_16x16x32_bf16(false, a.v, false, bc1.v,
                                                    (short)0, c1,  false, false);
      hh0 = __builtin_amdgcn_wmma_f32_16x16x32_bf16(false, a.v, false, bh0.v,
                                                    (short)0, hh0, false, false);
      hh1 = __builtin_amdgcn_wmma_f32_16x16x32_bf16(false, a.v, false, bh1.v,
                                                    (short)0, hh1, false, false);
    }
    __syncthreads();                           // reads of `cur` done before reuse
  }
}

// ---------------------------------------------------------------------------
// One full NAS-cell timestep. s and acc live in registers the whole kernel.
// On entry Abuf = [bf16(x_t) | bf16(h_prev)]; on exit h_out written and Abuf
// holds [bf16(x_{t+1}) | bf16(h_t)] for the next timestep.
// ---------------------------------------------------------------------------
__global__ __launch_bounds__(BLOCK_THREADS, 1) void nas_cell_step(
    const float* __restrict__ h_prev,   // [256,1024] fp32 (gate epilogue 0)
    const float* __restrict__ x_next,   // [256,1024] or null at t = L-1
    float* __restrict__ h_out,          // [256,1024]
    float* __restrict__ h_final,        // null except t = L-1
    const __bf16* __restrict__ W0T,     // [2048 n][2048 k]
    const __bf16* __restrict__ WsT,     // [8][2048 n][1024 k]
    __bf16* __restrict__ Abuf,          // [256][2048]  stage-0 A operand
    __bf16* __restrict__ Sbuf,          // [256][1024]  step A operand
    unsigned int* __restrict__ bar) {
  __shared__ __align__(16) unsigned char smem[2 * BUF_BYTES];

  const int lane = threadIdx.x & 31;
  const int wave = threadIdx.x >> 5;
  const int bm   = blockIdx.x & 1;            // 0..1 -> 128-row slab
  const int bj   = (int)blockIdx.x >> 1;      // 0..31 -> 32-col j-strip
  const int m0   = bm * 128 + wave * 16;
  const int j0   = bj * 32;
  const int l15  = lane & 15;
  const int rhi  = (lane >> 4) * 8;           // C/D row group offset

  v8f c0, c1, hh0, hh1;
  v8f s0, s1;
  v8f sum0 = v8f{}, sum1 = v8f{};

  // ---- stage 0: [x|h] @ W0, gate against h_prev --------------------------
  gemm_stage(Abuf, H2, H2, W0T, smem, j0, m0, c0, c1, hh0, hh1);
  {
    const float* hp = h_prev + (size_t)(m0 + rhi) * H + j0 + l15;
#pragma unroll
    for (int i = 0; i < 8; ++i) {
      float hpa = hp[(size_t)i * H];
      float hpb = hp[(size_t)i * H + 16];
      float sa  = hpa + sigmoid_f(c0[i]) * (tanhf(hh0[i]) - hpa);
      float sb  = hpb + sigmoid_f(c1[i]) * (tanhf(hh1[i]) - hpb);
      s0[i] = sa;
      s1[i] = sb;
      size_t r = (size_t)(m0 + rhi + i) * H + j0 + l15;
      Sbuf[r]      = to_bf16(sa);
      Sbuf[r + 16] = to_bf16(sb);
    }
  }
  grid_sync(bar);

  // ---- 8 NAS steps: s @ Ws[st], register-resident recurrence -------------
  for (int st = 0; st < STEPS; ++st) {
    gemm_stage(Sbuf, H, H, WsT + (size_t)st * H2 * H, smem, j0, m0,
               c0, c1, hh0, hh1);
    if (st < STEPS - 1) {
#pragma unroll
      for (int i = 0; i < 8; ++i) {
        float sa = s0[i], sb = s1[i];
        sa += sigmoid_f(c0[i]) * (tanhf(hh0[i]) - sa);
        sb += sigmoid_f(c1[i]) * (tanhf(hh1[i]) - sb);
        s0[i] = sa;
        s1[i] = sb;
        sum0[i] += sa;
        sum1[i] += sb;
        size_t r = (size_t)(m0 + rhi + i) * H + j0 + l15;
        Sbuf[r]      = to_bf16(sa);
        Sbuf[r + 16] = to_bf16(sb);
      }
      grid_sync(bar);
    } else {
      // final step: emit h = mean(states), seed Abuf for timestep t+1
#pragma unroll
      for (int i = 0; i < 8; ++i) {
        float sa = s0[i], sb = s1[i];
        sa += sigmoid_f(c0[i]) * (tanhf(hh0[i]) - sa);
        sb += sigmoid_f(c1[i]) * (tanhf(hh1[i]) - sb);
        float ha = (sum0[i] + sa) * (1.0f / STEPS);
        float hb = (sum1[i] + sb) * (1.0f / STEPS);
        size_t row = (size_t)(m0 + rhi + i);
        h_out[row * H + j0 + l15]      = ha;
        h_out[row * H + j0 + 16 + l15] = hb;
        if (h_final) {
          h_final[row * H + j0 + l15]      = ha;
          h_final[row * H + j0 + 16 + l15] = hb;
        }
        Abuf[row * H2 + H + j0 + l15]      = to_bf16(ha);
        Abuf[row * H2 + H + j0 + 16 + l15] = to_bf16(hb);
      }
      if (x_next) {  // convert x_{t+1} into Abuf[:, 0:H]
        const int tid = (int)blockIdx.x * BLOCK_THREADS + (int)threadIdx.x;
        for (int idx = tid; idx < NH; idx += GRID_BLOCKS * BLOCK_THREADS) {
          int n = idx >> 10, j = idx & (H - 1);
          Abuf[(size_t)n * H2 + j] = to_bf16(x_next[idx]);
        }
      }
    }
  }
}

// ---------------------------------------------------------------------------
// Per-launch setup: weights -> bf16 transposed [n][k]; Abuf = [bf16(x_0)|bf16(h)];
// reset grid barrier.
// ---------------------------------------------------------------------------
__global__ __launch_bounds__(256) void convert_setup(
    const float* __restrict__ inputs,   // [64,256,1024] (x_0 slice used)
    const float* __restrict__ hidden,   // [256,1024]
    const float* __restrict__ W0,       // [2048 k][2048 n]
    const float* __restrict__ Ws,       // [8][1024 k][2048 n]
    __bf16* __restrict__ W0T, __bf16* __restrict__ WsT,
    __bf16* __restrict__ Abuf, unsigned int* __restrict__ bar) {
  if (blockIdx.x == 0 && threadIdx.x == 0) *bar = 0u;
  const int nt  = gridDim.x * blockDim.x;
  const int tid = blockIdx.x * blockDim.x + threadIdx.x;

  const int T0 = H2 * H2;
  for (int idx = tid; idx < T0; idx += nt) {
    int n = idx >> 11, k = idx & (H2 - 1);
    W0T[idx] = to_bf16(W0[(size_t)k * H2 + n]);
  }
  const int T1 = STEPS * H2 * H;
  for (int idx = tid; idx < T1; idx += nt) {
    int i = idx / (H2 * H);
    int r = idx - i * (H2 * H);
    int n = r >> 10, k = r & (H - 1);
    WsT[idx] = to_bf16(Ws[(size_t)i * H * H2 + (size_t)k * H2 + n]);
  }
  for (int idx = tid; idx < NH; idx += nt) {
    int n = idx >> 10, j = idx & (H - 1);
    Abuf[(size_t)n * H2 + j]     = to_bf16(inputs[idx]);
    Abuf[(size_t)n * H2 + H + j] = to_bf16(hidden[idx]);
  }
}

// ---------------------------------------------------------------------------
extern "C" void kernel_launch(void* const* d_in, const int* in_sizes, int n_in,
                              void* d_out, int out_size, void* d_ws,
                              size_t ws_size, hipStream_t stream) {
  const float* inputs = (const float*)d_in[0];  // [64,256,1024]
  const float* hidden = (const float*)d_in[1];  // [256,1024]
  const float* W0     = (const float*)d_in[2];  // [2048,2048]
  const float* Ws     = (const float*)d_in[3];  // [8,1024,2048]
  float* out = (float*)d_out;  // hiddens[64,256,1024] ++ h_final[256,1024]

  // Workspace (~42 MiB), 256B-aligned sections.
  char* ws = (char*)d_ws;
  unsigned int* bar = (unsigned int*)ws;                 // 256 B
  __bf16* W0T  = (__bf16*)(ws + 256);                    // 8 MiB
  __bf16* WsT  = W0T + (size_t)H2 * H2;                  // 32 MiB
  __bf16* Abuf = WsT + (size_t)STEPS * H2 * H;           // 1 MiB
  __bf16* Sbuf = Abuf + (size_t)NB * H2;                 // 0.5 MiB

  convert_setup<<<2048, 256, 0, stream>>>(inputs, hidden, W0, Ws,
                                          W0T, WsT, Abuf, bar);

  for (int t = 0; t < SEQL; ++t) {
    const float* hp = (t == 0) ? hidden : out + (size_t)(t - 1) * NH;
    const float* xn = (t < SEQL - 1) ? inputs + (size_t)(t + 1) * NH : nullptr;
    float* ho = out + (size_t)t * NH;
    float* hf = (t == SEQL - 1) ? out + (size_t)SEQL * NH : nullptr;
    nas_cell_step<<<GRID_BLOCKS, BLOCK_THREADS, 0, stream>>>(
        hp, xn, ho, hf, W0T, WsT, Abuf, Sbuf, bar);
  }
}
